// GraphSelfAttentionLayer_19430432047704
// MI455X (gfx1250) — matrix operand order repaired and checked by reading, not compiled, not run on hardware
//
#include <hip/hip_runtime.h>
#include <hip/hip_bf16.h>

// B=8, N=1024, F=1024, H=8, HD=128
typedef __attribute__((ext_vector_type(16))) _Float16 v16h;
typedef __attribute__((ext_vector_type(8)))  float    v8f;
typedef __attribute__((ext_vector_type(4)))  unsigned int u32x4;
typedef __attribute__((ext_vector_type(8)))  int i32x8;
typedef __attribute__((ext_vector_type(4)))  int i32x4;

#define NTOK 1024
#define NF   1024
#define NH   8
#define HD   128

#if __has_builtin(__builtin_amdgcn_tensor_load_to_lds)
#define HAVE_TDM 1
#else
#define HAVE_TDM 0
#endif

__device__ __forceinline__ v8f wmma16(v16h a, v16h b, v8f c) {
    return __builtin_amdgcn_wmma_f32_16x16x32_f16(
        false, a, false, b, (short)0, c, false, false);
}

// A-matrix (16x32 f16) K map: lanes 0-15 hold K 0..7 / 16..23, lanes 16-31 +8.
__device__ __forceinline__ int kmapA(int lane, int v) {
    return ((v >> 2) << 4) + (((lane >> 4) & 1) << 3) + ((v & 3) << 1);
}
// B-matrix (32x16 f16) K map: same as A: lanes 0-15 hold K 0..7 / 16..23, lanes 16-31 +8.
__device__ __forceinline__ int kmapB(int lane, int v) {
    return ((v >> 2) << 4) + (((lane >> 4) & 1) << 3) + ((v & 3) << 1);
}

__device__ __forceinline__ v16h frag_a(const _Float16* base, int lane, int ld) {
    const _Float16* p = base + (size_t)(lane & 15) * ld;
    v16h f;
#pragma unroll
    for (int v = 0; v < 8; ++v) {
        int k = kmapA(lane, v);
        f[2 * v] = p[k]; f[2 * v + 1] = p[k + 1];
    }
    return f;
}
__device__ __forceinline__ v16h frag_b(const _Float16* base, int lane, int ld) {
    const _Float16* p = base + (size_t)(lane & 15) * ld;
    v16h f;
#pragma unroll
    for (int v = 0; v < 8; ++v) {
        int k = kmapB(lane, v);
        f[2 * v] = p[k]; f[2 * v + 1] = p[k + 1];
    }
    return f;
}

// ---------------------------------------------------------------------------
// TDM 2-D f16 tile load (D# per CDNA5 ISA 08_async_tensor).
// ---------------------------------------------------------------------------
__device__ __forceinline__ void tdm_load_2d_f16(
    unsigned lds_off, const _Float16* gptr,
    unsigned tw, unsigned th, unsigned stride_elems) {
#if HAVE_TDM
    unsigned long long ga = (unsigned long long)(size_t)gptr;
    u32x4 g0 = { 1u, lds_off, (unsigned)ga,
                 (unsigned)((ga >> 32) & 0x01FFFFFFu) | 0x80000000u };
    i32x8 g1;
    g1[0] = (int)(1u << 16);                                  // data_size=2B
    g1[1] = (int)((stride_elems & 0xffffu) << 16);            // tensor_dim0 lo
    g1[2] = (int)((stride_elems >> 16) & 0xffffu)             // tensor_dim0 hi
          | (int)((th & 0xffffu) << 16);                      // tensor_dim1 lo
    g1[3] = (int)(tw << 16);                                  // tile_dim0
    g1[4] = (int)th;                                          // tile_dim1
    g1[5] = (int)stride_elems;                                // dim0_stride lo
    g1[6] = 0; g1[7] = 0;
    i32x4 z4 = {0, 0, 0, 0};
#if __clang_major__ >= 23
    i32x8 z8 = {0, 0, 0, 0, 0, 0, 0, 0};
    __builtin_amdgcn_tensor_load_to_lds(g0, g1, z4, z4, z8, 0);
#else
    __builtin_amdgcn_tensor_load_to_lds(g0, g1, z4, z4, 0);
#endif
#endif
}

__device__ __forceinline__ void tdm_wait(int n) {
#if HAVE_TDM
    if (n == 0)      __builtin_amdgcn_s_wait_tensorcnt((short)0);
    else if (n == 1) __builtin_amdgcn_s_wait_tensorcnt((short)1);
    else             __builtin_amdgcn_s_wait_tensorcnt((short)2);
#endif
}

__device__ __forceinline__ void stage_tile_f16(
    _Float16* lds, const _Float16* g,
    unsigned tw, unsigned th, unsigned stride, int t, int nt) {
#if HAVE_TDM
    (void)t; (void)nt;
    tdm_load_2d_f16((unsigned)(size_t)lds, g, tw, th, stride);
#else
    int total = (int)(tw * th) >> 1;
    for (int i = t; i < total; i += nt) {
        int r = (2 * i) / (int)tw, c = (2 * i) % (int)tw;
        ((unsigned*)lds)[i] = *(const unsigned*)(g + (size_t)r * stride + c);
    }
#endif
}

// ---------------------------------------------------------------------------
// fp32 -> f16 conversion
// ---------------------------------------------------------------------------
__global__ __launch_bounds__(256) void cvt_f16(
    const float* __restrict__ s, _Float16* __restrict__ d, int n) {
    int i = (blockIdx.x * 256 + threadIdx.x) * 8;
    if (i < n) {
#pragma unroll
        for (int j = 0; j < 8; ++j) d[i + j] = (_Float16)s[i + j];
    }
}

// ---------------------------------------------------------------------------
// GEMM-NT f16: out[M,1024] = A[M,1024] @ W[64n,1024]^T (+bias / transposed).
// Block 256 thr, 64x64 tile; wave grid 4x2, each wave 16x32 (2 accumulators,
// A-fragment reused). K chunks of 32, TDM double-buffered.
// ---------------------------------------------------------------------------
__device__ __forceinline__ void gemm_tile_64x64(
    const _Float16* Ab, const _Float16* Wb, _Float16* As, _Float16* Ws,
    int tid, int lane, int wid, v8f& acc0, v8f& acc1) {
    const int wm = (wid & 3) * 16, wn = (wid >> 2) * 32;
#if HAVE_TDM
    if (wid == 0) {
        stage_tile_f16(As, Ab, 32, 64, NF, lane, 32);
        stage_tile_f16(Ws, Wb, 32, 64, NF, lane, 32);
    }
#else
    stage_tile_f16(As, Ab, 32, 64, NF, tid, 256);
    stage_tile_f16(Ws, Wb, 32, 64, NF, tid, 256);
#endif
    for (int kc = 0; kc < 32; ++kc) {
        const int cur = kc & 1, nxt = cur ^ 1;
        __syncthreads();  // prior reads of buf nxt complete
#if HAVE_TDM
        if (wid == 0) {
            if (kc < 31) {
                stage_tile_f16(As + nxt * 64 * 32, Ab + (kc + 1) * 32, 32, 64, NF, lane, 32);
                stage_tile_f16(Ws + nxt * 64 * 32, Wb + (kc + 1) * 32, 32, 64, NF, lane, 32);
                tdm_wait(2);
            } else {
                tdm_wait(0);
            }
        }
#else
        if (kc < 31) {
            stage_tile_f16(As + nxt * 64 * 32, Ab + (kc + 1) * 32, 32, 64, NF, tid, 256);
            stage_tile_f16(Ws + nxt * 64 * 32, Wb + (kc + 1) * 32, 32, 64, NF, tid, 256);
        }
#endif
        __syncthreads();  // buf cur ready
        v16h a  = frag_a(As + cur * 64 * 32 + wm * 32, lane, 32);
        v16h b0 = frag_b(Ws + cur * 64 * 32 + wn * 32, lane, 32);
        v16h b1 = frag_b(Ws + cur * 64 * 32 + (wn + 16) * 32, lane, 32);
        acc0 = wmma16(a, b0, acc0);
        acc1 = wmma16(a, b1, acc1);
    }
}

__global__ __launch_bounds__(256) void proj_f16(
    const _Float16* __restrict__ A, const _Float16* __restrict__ W,
    const float* __restrict__ bias, _Float16* __restrict__ out) {
    __shared__ _Float16 As[2 * 64 * 32];
    __shared__ _Float16 Ws[2 * 64 * 32];
    const int tid = threadIdx.x, lane = tid & 31, wid = tid >> 5;
    const int m0 = blockIdx.x * 64, n0 = blockIdx.y * 64;
    v8f acc0 = {}, acc1 = {};
    gemm_tile_64x64(A + (size_t)m0 * NF, W + (size_t)n0 * NF, As, Ws,
                    tid, lane, wid, acc0, acc1);
    const int wm = (wid & 3) * 16, wn = (wid >> 2) * 32;
    const int col0 = n0 + wn + (lane & 15), col1 = col0 + 16;
    const float bv0 = bias[col0], bv1 = bias[col1];
#pragma unroll
    for (int e = 0; e < 8; ++e) {
        int row = m0 + wm + e + 8 * (lane >> 4);
        out[(size_t)row * NF + col0] = (_Float16)(acc0[e] + bv0);
        out[(size_t)row * NF + col1] = (_Float16)(acc1[e] + bv1);
    }
}

// Vp (per b,h): Vp[m,d] = sum_f V[b,m,f]*Wo[h,d,f]; stored transposed [d][m]
__global__ __launch_bounds__(256) void vp_proj(
    const _Float16* __restrict__ V, const _Float16* __restrict__ Wo,
    _Float16* __restrict__ VpT) {
    __shared__ _Float16 As[2 * 64 * 32];
    __shared__ _Float16 Ws[2 * 64 * 32];
    const int tid = threadIdx.x, lane = tid & 31, wid = tid >> 5;
    const int bh = blockIdx.z, b = bh >> 3, h = bh & 7;
    const int m0 = blockIdx.x * 64, n0 = blockIdx.y * 64;
    v8f acc0 = {}, acc1 = {};
    gemm_tile_64x64(V + (size_t)b * NTOK * NF + (size_t)m0 * NF,
                    Wo + (size_t)h * HD * NF + (size_t)n0 * NF,
                    As, Ws, tid, lane, wid, acc0, acc1);
    _Float16* outT = VpT + (size_t)bh * HD * NTOK;
    const int wm = (wid & 3) * 16, wn = (wid >> 2) * 32;
    const int d0 = n0 + wn + (lane & 15), d1 = d0 + 16;
#pragma unroll
    for (int e = 0; e < 8; ++e) {
        int m = m0 + wm + e + 8 * (lane >> 4);
        outT[(size_t)d0 * NTOK + m] = (_Float16)acc0[e];
        outT[(size_t)d1 * NTOK + m] = (_Float16)acc1[e];
    }
}

// ---------------------------------------------------------------------------
// Attention: block = (b, 16-query tile), 8 waves cooperate per head.
// LDS: S f32, ATT f32, MB f32 (mask+bias once), Sh f16 (probs), Qs f16,
// per-wave K/Vp TDM double buffers.  Total ~293 KB of the 320 KB WGP LDS.
// ---------------------------------------------------------------------------
__global__ __launch_bounds__(256) void attn(
    const _Float16* __restrict__ Q, const _Float16* __restrict__ Kh,
    const _Float16* __restrict__ VpT, const int* __restrict__ adj,
    const float* __restrict__ lb, const float* __restrict__ bo,
    float* __restrict__ out, float* __restrict__ attavg) {
    extern __shared__ char smem[];
    float* S   = (float*)smem;                        // 16*1024 f32
    float* ATT = S + 16 * NTOK;                       // 16*1024 f32
    float* MB  = ATT + 16 * NTOK;                     // 16*1024 f32
    _Float16* Sh = (_Float16*)(MB + 16 * NTOK);       // 16*1024 f16 (probs)
    _Float16* Qs = Sh + 16 * NTOK;                    // 16*128 f16
    _Float16* Ks = Qs + 16 * HD;                      // 8 waves * 2 * 16*128
    float* red  = (float*)(Ks + 8 * 2 * 16 * HD);     // 256
    float* rmax = red + 256;                          // 16
    float* rsum = rmax + 16;                          // 16

    const int tid = threadIdx.x, lane = tid & 31, wid = tid >> 5;
    const int b = blockIdx.x >> 6;
    const int n0 = (blockIdx.x & 63) << 4;
    const float scale = 0.08838834764831845f;  // 1/sqrt(128)
    _Float16* Kw = Ks + wid * (2 * 16 * HD);

    // init ATT and head-independent mask+bias: MB = adj>0 ? lb : lb - 9e15
#pragma unroll 8
    for (int j = 0; j < 64; ++j) {
        int i = tid + j * 256;
        size_t gi = (size_t)(b * NTOK + n0 + (i >> 10)) * NTOK + (i & 1023);
        MB[i] = (adj[gi] > 0) ? lb[gi] : (lb[gi] - 9e15f);
        ATT[i] = 0.f;
    }

    for (int h = 0; h < NH; ++h) {
        __syncthreads();
        // stage Q tile [16 x 128] (dword copies)
        {
            const unsigned* qsrc =
                (const unsigned*)(Q + (size_t)(b * NTOK + n0) * NF + h * HD);
            unsigned* qd = (unsigned*)Qs;
#pragma unroll
            for (int j = 0; j < 4; ++j) {
                int i = tid + j * 256;
                qd[i] = qsrc[(size_t)(i >> 6) * (NF / 2) + (i & 63)];
            }
        }
        __syncthreads();
        v16h aq[4];
#pragma unroll
        for (int c = 0; c < 4; ++c) aq[c] = frag_a(Qs + c * 32, lane, HD);

        // ---- scores: each wave 8 key tiles, TDM double-buffered K staging
        const _Float16* Kbase = Kh + (size_t)(b * NTOK) * NF + h * HD;
        stage_tile_f16(Kw, Kbase + (size_t)(wid * 8) * 16 * NF, HD, 16, NF, lane, 32);
        for (int t = 0; t < 8; ++t) {
            const int cur = t & 1;
            if (t < 7)
                stage_tile_f16(Kw + (cur ^ 1) * 16 * HD,
                               Kbase + (size_t)(wid * 8 + t + 1) * 16 * NF,
                               HD, 16, NF, lane, 32);
#if HAVE_TDM
            tdm_wait(t < 7 ? 1 : 0);
#endif
            const _Float16* Kt = Kw + cur * 16 * HD;
            v8f acc = {};
#pragma unroll
            for (int c = 0; c < 4; ++c)
                acc = wmma16(aq[c], frag_b(Kt + c * 32, lane, HD), acc);
            const int m0 = (wid * 8 + t) * 16;
            const int col = m0 + (lane & 15);
#pragma unroll
            for (int e = 0; e < 8; ++e) {
                int row = e + 8 * (lane >> 4);
                float mb = MB[row * NTOK + col];
                float s = acc[e] * scale + mb;
                S[row * NTOK + col] = (mb < -1e14f) ? mb : s;
            }
        }
        __syncthreads();
        // ---- fp32 softmax over keys: 16 threads per query row; probs -> Sh f16
        {
            const int q = tid >> 4, t = tid & 15;
            float mx = -3.4e38f;
            for (int m = t; m < NTOK; m += 16) mx = fmaxf(mx, S[q * NTOK + m]);
            red[q * 16 + t] = mx;
            __syncthreads();
            if (t == 0) {
                float m2 = red[q * 16];
                for (int i = 1; i < 16; ++i) m2 = fmaxf(m2, red[q * 16 + i]);
                rmax[q] = m2;
            }
            __syncthreads();
            const float m2 = rmax[q];
            float sm = 0.f;
            for (int m = t; m < NTOK; m += 16) {
                float e = __expf(S[q * NTOK + m] - m2);
                S[q * NTOK + m] = e;
                sm += e;
            }
            red[q * 16 + t] = sm;
            __syncthreads();
            if (t == 0) {
                float s2 = 0.f;
                for (int i = 0; i < 16; ++i) s2 += red[q * 16 + i];
                rsum[q] = 1.f / s2;
            }
            __syncthreads();
            const float inv = rsum[q];
            for (int m = t; m < NTOK; m += 16) {
                float a = S[q * NTOK + m] * inv;
                Sh[q * NTOK + m] = (_Float16)a;   // f16 probs, converted ONCE
                ATT[q * NTOK + m] += a;
            }
        }
        __syncthreads();
        // ---- O = A @ Vp_h: wave wid -> d-subtile at d0, TDM-staged Vp chunks
        const int d0 = wid * 16;
        const _Float16* Vb =
            VpT + (size_t)(b * NH + h) * HD * NTOK + (size_t)d0 * NTOK;
        _Float16* Vs = Kw;  // reuse per-wave buffers (2 x 16x32)
        stage_tile_f16(Vs, Vb, 32, 16, NTOK, lane, 32);
        v8f acc = {};
        for (int c = 0; c < 32; ++c) {
            const int cur = c & 1;
            if (c < 31)
                stage_tile_f16(Vs + (cur ^ 1) * 16 * 32, Vb + (c + 1) * 32,
                               32, 16, NTOK, lane, 32);
#if HAVE_TDM
            tdm_wait(c < 31 ? 1 : 0);
#endif
            v16h a = frag_a(Sh + c * 32, lane, NTOK);
            v16h bb = frag_b(Vs + cur * 16 * 32, lane, 32);
            acc = wmma16(a, bb, acc);
        }
        const int dcol = d0 + (lane & 15);
        const float bov = bo[h * HD + dcol];
#pragma unroll
        for (int e = 0; e < 8; ++e) {
            int row = e + 8 * (lane >> 4);
            out[(size_t)(b * NTOK + n0 + row) * NF + h * HD + dcol] = acc[e] + bov;
        }
    }
    __syncthreads();
#pragma unroll 8
    for (int j = 0; j < 64; ++j) {
        int i = tid + j * 256;
        attavg[(size_t)(b * NTOK + n0 + (i >> 10)) * NTOK + (i & 1023)] =
            ATT[i] * 0.125f;
    }
}

extern "C" void kernel_launch(void* const* d_in, const int* in_sizes, int n_in,
                              void* d_out, int out_size, void* d_ws, size_t ws_size,
                              hipStream_t stream) {
    const float* obj   = (const float*)d_in[0];
    const float* cross = (const float*)d_in[1];
    const int*   adj   = (const int*)d_in[2];
    const float* lb    = (const float*)d_in[3];
    const float* Wq    = (const float*)d_in[4];
    const float* bq    = (const float*)d_in[5];
    const float* Wk    = (const float*)d_in[6];
    const float* bk    = (const float*)d_in[7];
    const float* Wv    = (const float*)d_in[8];
    const float* bv    = (const float*)d_in[9];
    const float* Wo    = (const float*)d_in[10];
    const float* bo    = (const float*)d_in[11];

    const size_t big = 8ull * 1024 * 1024;   // 8M elems
    const size_t wsz = 1024ull * 1024;       // 1M elems
    _Float16* Qh  = (_Float16*)d_ws;
    _Float16* Kh  = Qh + big;
    _Float16* Vh  = Kh + big;
    _Float16* VpT = Vh + big;
    _Float16* Xo  = VpT + big;
    _Float16* Xc  = Xo + big;
    _Float16* Wqh = Xc + big;
    _Float16* Wkh = Wqh + wsz;
    _Float16* Wvh = Wkh + wsz;
    _Float16* Woh = Wvh + wsz;

    dim3 blk(256);
    cvt_f16<<<4096, blk, 0, stream>>>(obj,   Xo,  (int)big);
    cvt_f16<<<4096, blk, 0, stream>>>(cross, Xc,  (int)big);
    cvt_f16<<<512,  blk, 0, stream>>>(Wq,    Wqh, (int)wsz);
    cvt_f16<<<512,  blk, 0, stream>>>(Wk,    Wkh, (int)wsz);
    cvt_f16<<<512,  blk, 0, stream>>>(Wv,    Wvh, (int)wsz);
    cvt_f16<<<512,  blk, 0, stream>>>(Wo,    Woh, (int)wsz);

    dim3 pg(128, 16);  // 8192/64 x 1024/64
    proj_f16<<<pg, blk, 0, stream>>>(Xo, Wqh, bq, Qh);
    proj_f16<<<pg, blk, 0, stream>>>(Xc, Wkh, bk, Kh);
    proj_f16<<<pg, blk, 0, stream>>>(Xc, Wvh, bv, Vh);
    vp_proj<<<dim3(16, 2, 64), blk, 0, stream>>>(Vh, Woh, VpT);

    size_t smem = (size_t)(16 * 1024 * 3) * sizeof(float)       // S + ATT + MB
                + (size_t)(16 * 1024) * sizeof(_Float16)        // Sh (f16 probs)
                + (size_t)(16 * 128) * sizeof(_Float16)         // Qs
                + (size_t)(8 * 2 * 16 * 128) * sizeof(_Float16) // K/Vp buffers
                + (size_t)(256 + 32) * sizeof(float);           // reductions
    float* out_ptr = (float*)d_out;
    float* att_ptr = out_ptr + 8ull * 1024 * 1024;
    attn<<<dim3(512), blk, smem, stream>>>(Qh, Kh, VpT, adj, lb, bo,
                                           out_ptr, att_ptr);
}
